// AttentionBlock_34832184771292
// MI455X (gfx1250) — compile-verified, matrix operand order checked by reading, not run on hardware
//
#include <hip/hip_runtime.h>
#include <cstdint>

typedef uint32_t u32;
typedef uint64_t u64;
typedef _Float16 f16;
typedef __attribute__((ext_vector_type(16))) _Float16 v16h;
typedef __attribute__((ext_vector_type(8)))  float    v8f;
typedef uint32_t u32x4 __attribute__((ext_vector_type(4)));
typedef int      i32x8 __attribute__((ext_vector_type(8)));
typedef int      i32x4 __attribute__((ext_vector_type(4)));

// ---------------------------------------------------------------------------
// TDM: issue a 2D f16 tile load (global -> LDS) via Tensor Data Mover.
// D# group0/group1 packed per CDNA5 ISA 8.3/8.4. tensor dims == tile dims
// (all our tiles are in-bounds), optional LDS padding for bank-conflict-free
// row strides. Wave-level op: call from exactly one wave.
// ---------------------------------------------------------------------------
__device__ __forceinline__ void tdm_load_tile_f16(
    u32 lds_byte_off, const f16* gptr,
    u32 tile_d0, u32 tile_d1, u64 stride0_elems,
    u32 pad_interval_code, u32 pad_amount_code, bool pad_en)
{
  u64 ga = (u64)(const void*)gptr;
  u32x4 g0;
  g0[0] = 1u;                                   // count=1, user descriptor
  g0[1] = lds_byte_off;                         // LDS dest (bytes)
  g0[2] = (u32)ga;                              // global_addr[31:0]
  g0[3] = ((u32)(ga >> 32) & 0x01FFFFFFu) | (2u << 30);  // [56:32] | type=2
  u32 w0 = (1u << 16)                           // data_size = 2 bytes (f16)
         | (pad_en ? (1u << 20) : 0u)
         | (pad_interval_code << 22)
         | (pad_amount_code << 25);
  u32 td0 = tile_d0, td1 = tile_d1;
  u32 w1 = (td0 & 0xFFFFu) << 16;               // tensor_dim0[15:0]
  u32 w2 = (td0 >> 16) | ((td1 & 0xFFFFu) << 16);
  u32 w3 = (td1 >> 16) | ((tile_d0 & 0xFFFFu) << 16);   // tile_dim0
  u32 w4 = (tile_d1 & 0xFFFFu);                 // tile_dim1 (tile_dim2 = 0)
  u32 w5 = (u32)stride0_elems;                  // tensor_dim0_stride[31:0]
  u32 w6 = (u32)((stride0_elems >> 32) & 0xFFFFu);
  i32x8 g1 = { (int)w0, (int)w1, (int)w2, (int)w3,
               (int)w4, (int)w5, (int)w6, 0 };
  i32x4 gz4 = { 0, 0, 0, 0 };
  i32x8 gz8 = { 0, 0, 0, 0, 0, 0, 0, 0 };
  __builtin_amdgcn_tensor_load_to_lds(g0, g1, gz4, gz4, gz8, 0);
}

// ---------------------------------------------------------------------------
// WMMA fragment loaders (wave32, 16-bit A layout per ISA 7.12.2)
// ---------------------------------------------------------------------------
// Rows along M (or N for transposed-B), contiguous K in LDS.
// lane<16: rows 0..15, K base 0 ; lane>=16: same rows, K base +8.
__device__ __forceinline__ v16h load_frag_k(const f16* t, int row0, int rstride, int lane)
{
  int r  = row0 + (lane & 15);
  int kk = (lane >> 4) << 3;
  const f16* p = t + r * rstride + kk;
  v16h v;
#pragma unroll
  for (int i = 0; i < 4; ++i) {
    v[2 * i]     = p[2 * i];
    v[2 * i + 1] = p[2 * i + 1];
    v[8 + 2 * i]     = p[16 + 2 * i];
    v[8 + 2 * i + 1] = p[16 + 2 * i + 1];
  }
  return v;
}

// B stored [K][N] row-major in LDS: gather one column per lane.
__device__ __forceinline__ v16h load_frag_kn(const f16* t, int n0, int rstride, int lane)
{
  int n  = n0 + (lane & 15);
  int kb = (lane >> 4) << 3;
  v16h v;
#pragma unroll
  for (int i = 0; i < 4; ++i) {
    v[2 * i]         = t[(kb + 2 * i) * rstride + n];
    v[2 * i + 1]     = t[(kb + 2 * i + 1) * rstride + n];
    v[8 + 2 * i]     = t[(kb + 16 + 2 * i) * rstride + n];
    v[8 + 2 * i + 1] = t[(kb + 16 + 2 * i + 1) * rstride + n];
  }
  return v;
}

__device__ __forceinline__ v8f wmma_f16(v16h a, v16h b, v8f c)
{
  return __builtin_amdgcn_wmma_f32_16x16x32_f16(
      false, a, false, b, (short)0, c, false, false);
}

// ---------------------------------------------------------------------------
// GroupNorm: x[B,C,HW] f32 -> hf[B,HW,C] f16. One block per (batch, group).
// ---------------------------------------------------------------------------
__global__ __launch_bounds__(256)
void gn_kernel(const float* __restrict__ x, const float* __restrict__ w,
               const float* __restrict__ b, f16* __restrict__ hf)
{
  const int C = 512, HW = 1024, G = 32, CPG = 16;
  int bb = blockIdx.x / G, g = blockIdx.x % G;
  const float* xb = x + ((size_t)bb * C + (size_t)g * CPG) * HW;
  __shared__ float s1[256], s2[256];
  float sum = 0.f, ssq = 0.f;
  for (int i = threadIdx.x; i < CPG * HW; i += 256) {
    float v = xb[i];
    sum += v; ssq += v * v;
  }
  s1[threadIdx.x] = sum; s2[threadIdx.x] = ssq;
  __syncthreads();
  for (int s = 128; s > 0; s >>= 1) {
    if (threadIdx.x < s) {
      s1[threadIdx.x] += s1[threadIdx.x + s];
      s2[threadIdx.x] += s2[threadIdx.x + s];
    }
    __syncthreads();
  }
  const float inv_n = 1.0f / (CPG * HW);
  float mean = s1[0] * inv_n;
  float var  = s2[0] * inv_n - mean * mean;
  float rstd = rsqrtf(var + 1e-5f);
  for (int i = threadIdx.x; i < CPG * HW; i += 256) {
    int cl = i >> 10, p = i & 1023;
    int c = g * CPG + cl;
    float v = (xb[i] - mean) * rstd * w[c] + b[c];
    hf[((size_t)bb * HW + p) * C + c] = (f16)v;
  }
}

__global__ __launch_bounds__(256)
void cvt_f32_to_f16(const float* __restrict__ s, f16* __restrict__ d, int n)
{
  int i = blockIdx.x * 256 + threadIdx.x;
  if (i < n) d[i] = (f16)s[i];
}

// ---------------------------------------------------------------------------
// Row softmax in place on f16 scores: rows of length 1024, one block per row.
// ---------------------------------------------------------------------------
__global__ __launch_bounds__(256)
void softmax_kernel(f16* __restrict__ s)
{
  f16* r = s + (size_t)blockIdx.x * 1024;
  int t = threadIdx.x;
  __shared__ float red[256];
  float v[4];
  float mx = -1e30f;
#pragma unroll
  for (int i = 0; i < 4; ++i) { v[i] = (float)r[t + i * 256]; mx = fmaxf(mx, v[i]); }
  red[t] = mx; __syncthreads();
  for (int st = 128; st > 0; st >>= 1) {
    if (t < st) red[t] = fmaxf(red[t], red[t + st]);
    __syncthreads();
  }
  mx = red[0];
  __syncthreads();
  float sum = 0.f;
#pragma unroll
  for (int i = 0; i < 4; ++i) { v[i] = __expf(v[i] - mx); sum += v[i]; }
  red[t] = sum; __syncthreads();
  for (int st = 128; st > 0; st >>= 1) {
    if (t < st) red[t] += red[t + st];
    __syncthreads();
  }
  float rinv = 1.0f / red[0];
#pragma unroll
  for (int i = 0; i < 4; ++i) r[t + i * 256] = (f16)(v[i] * rinv);
}

// ---------------------------------------------------------------------------
// Tiled WMMA GEMM:  D[M,N] = alpha * A[M,K] x op(B) + bias
//   TRANSB=1: B stored [N][K] row-major (weights [O][C], or K for Q.K^T)
//   TRANSB=0: B stored [K][N] row-major (V for P.V)
//   RESID=1:  f32 epilogue: out[b, n, p] = acc + bias[n] + resid[b, n, p]
//             (transposed store back to [B, C, H, W]); else f16 row-major.
// Block: 256 thr = 8 waves, block tile 128x128, wave tile 32x64, K-step 32.
// Tiles staged global->LDS by TDM (wave 0), double-buffered, TENSORcnt sync.
// ---------------------------------------------------------------------------
template <int TRANSB, int RESID>
__global__ __launch_bounds__(256)
void gemm_wmma_kernel(const f16* __restrict__ A, const f16* __restrict__ B,
                      f16* __restrict__ outH, float* __restrict__ outF,
                      const float* __restrict__ bias, const float* __restrict__ resid,
                      int M, int N, int K,
                      long long sA, long long sB, long long sOut,
                      float alpha)
{
  constexpr int AST = 36;                       // A row stride (halves), TDM-padded
  constexpr int BST = TRANSB ? 36 : 132;        // B row stride (halves)
  constexpr int ASZ = 128 * 36;
  constexpr int BSZ = TRANSB ? 128 * 36 : 32 * 132;
  __shared__ f16 As[2][ASZ];
  __shared__ f16 Bs[2][BSZ];

  const int n0 = blockIdx.x * 128;
  const int m0 = blockIdx.y * 128;
  const int bz = blockIdx.z;
  const f16* Ab = A + (size_t)bz * sA;
  const f16* Bb = B + (size_t)bz * sB;

  const int tid  = threadIdx.x;
  const int wave = tid >> 5;
  const int lane = tid & 31;
  const int wm = (wave & 3) * 32;               // 4 waves along M
  const int wn = (wave >> 2) * 64;              // 2 waves along N
  const bool isW0 = (wave == 0);

  const int nk = K >> 5;                        // K / 32

  const u32 asBase = (u32)(u64)(void*)&As[0][0];
  const u32 bsBase = (u32)(u64)(void*)&Bs[0][0];

  auto issue = [&](int kt, int buf) {
    // A tile: [128 rows of M][32 K], row = 16 DWORDs -> pad interval code 3,
    // amount code 1 (2 DWORDs) => LDS row stride 36 halves.
    tdm_load_tile_f16(asBase + (u32)buf * (ASZ * 2),
                      Ab + (size_t)m0 * K + (size_t)kt * 32,
                      32, 128, (u64)K, 3, 1, true);
    if (TRANSB) {
      tdm_load_tile_f16(bsBase + (u32)buf * (BSZ * 2),
                        Bb + (size_t)n0 * K + (size_t)kt * 32,
                        32, 128, (u64)K, 3, 1, true);
    } else {
      // B tile: [32 rows of K][128 N], row = 64 DWORDs -> pad interval code 5,
      // amount code 1 => LDS row stride 132 halves.
      tdm_load_tile_f16(bsBase + (u32)buf * (BSZ * 2),
                        Bb + (size_t)kt * 32 * N + n0,
                        128, 32, (u64)N, 5, 1, true);
    }
  };

  const v8f vzero = {0.f, 0.f, 0.f, 0.f, 0.f, 0.f, 0.f, 0.f};
  v8f acc[2][4];
#pragma unroll
  for (int mi = 0; mi < 2; ++mi)
#pragma unroll
    for (int ni = 0; ni < 4; ++ni) acc[mi][ni] = vzero;

  if (isW0) issue(0, 0);

  for (int kt = 0; kt < nk; ++kt) {
    const int buf = kt & 1;
    if (isW0) __builtin_amdgcn_s_wait_tensorcnt(0);   // tiles for kt landed
    __syncthreads();                                  // release all waves
    if (isW0 && (kt + 1) < nk) issue(kt + 1, buf ^ 1);

    v16h a0 = load_frag_k(&As[buf][0], wm + 0,  AST, lane);
    v16h a1 = load_frag_k(&As[buf][0], wm + 16, AST, lane);
    v16h b0, b1, b2, b3;
    if (TRANSB) {
      b0 = load_frag_k(&Bs[buf][0], wn + 0,  BST, lane);
      b1 = load_frag_k(&Bs[buf][0], wn + 16, BST, lane);
      b2 = load_frag_k(&Bs[buf][0], wn + 32, BST, lane);
      b3 = load_frag_k(&Bs[buf][0], wn + 48, BST, lane);
    } else {
      b0 = load_frag_kn(&Bs[buf][0], wn + 0,  BST, lane);
      b1 = load_frag_kn(&Bs[buf][0], wn + 16, BST, lane);
      b2 = load_frag_kn(&Bs[buf][0], wn + 32, BST, lane);
      b3 = load_frag_kn(&Bs[buf][0], wn + 48, BST, lane);
    }
    acc[0][0] = wmma_f16(a0, b0, acc[0][0]);
    acc[0][1] = wmma_f16(a0, b1, acc[0][1]);
    acc[0][2] = wmma_f16(a0, b2, acc[0][2]);
    acc[0][3] = wmma_f16(a0, b3, acc[0][3]);
    acc[1][0] = wmma_f16(a1, b0, acc[1][0]);
    acc[1][1] = wmma_f16(a1, b1, acc[1][1]);
    acc[1][2] = wmma_f16(a1, b2, acc[1][2]);
    acc[1][3] = wmma_f16(a1, b3, acc[1][3]);
    __syncthreads();                                  // reads done before overwrite
  }

  // Epilogue. C/D layout: VGPR j -> M = j + 8*(lane>=16), N = lane&15.
  const int hh = lane >> 4;
  const int nn = lane & 15;
#pragma unroll
  for (int mi = 0; mi < 2; ++mi) {
#pragma unroll
    for (int ni = 0; ni < 4; ++ni) {
      int mbase = m0 + wm + mi * 16 + 8 * hh;
      int nbase = n0 + wn + ni * 16 + nn;
      float bv = (bias != nullptr) ? bias[nbase] : 0.f;
#pragma unroll
      for (int j = 0; j < 8; ++j) {
        int m = mbase + j;
        float val = acc[mi][ni][j] * alpha + bv;
        if (RESID) {
          // global row m = b*1024 + p ; out[b][c=nbase][p] (+ residual x)
          size_t o = (size_t)(m >> 10) * (512u * 1024u)
                   + (size_t)nbase * 1024u + (size_t)(m & 1023);
          outF[o] = val + resid[o];
        } else {
          outH[(size_t)bz * sOut + (size_t)m * N + nbase] = (f16)val;
        }
      }
    }
  }
}

// ---------------------------------------------------------------------------
// Host-side orchestration
// ---------------------------------------------------------------------------
extern "C" void kernel_launch(void* const* d_in, const int* in_sizes, int n_in,
                              void* d_out, int out_size, void* d_ws, size_t ws_size,
                              hipStream_t stream)
{
  (void)in_sizes; (void)n_in; (void)out_size; (void)ws_size;
  const float* x  = (const float*)d_in[0];
  const float* nw = (const float*)d_in[1];
  const float* nb = (const float*)d_in[2];
  const float* wq = (const float*)d_in[3];  const float* bq = (const float*)d_in[4];
  const float* wk = (const float*)d_in[5];  const float* bk = (const float*)d_in[6];
  const float* wv = (const float*)d_in[7];  const float* bv = (const float*)d_in[8];
  const float* wo = (const float*)d_in[9];  const float* bo = (const float*)d_in[10];
  float* out = (float*)d_out;

  // Workspace layout (bytes)
  char* ws = (char*)d_ws;
  f16* hf   = (f16*)(ws);                          // [16,1024,512]   16 MB
  f16* wq16 = (f16*)(ws + 16777216);               // [512,512]       0.5 MB
  f16* wk16 = wq16 + 262144;
  f16* wv16 = wq16 + 524288;
  f16* wo16 = wq16 + 786432;
  f16* q16  = (f16*)(ws + 18874368);               // [16,1024,512]   16 MB
  f16* k16  = q16 + 8388608;
  f16* v16  = q16 + 16777216;
  f16* p16  = (f16*)(ws + 69206016);               // [16,1024,1024]  32 MB
  f16* o16  = (f16*)(ws + 102760448);              // [16,1024,512]   16 MB

  // 1) GroupNorm -> f16 [B, HW, C]
  gn_kernel<<<512, 256, 0, stream>>>(x, nw, nb, hf);

  // 2) Weights f32 -> f16 (keep [O][C]; consumed with TRANSB GEMM)
  cvt_f32_to_f16<<<1024, 256, 0, stream>>>(wq, wq16, 262144);
  cvt_f32_to_f16<<<1024, 256, 0, stream>>>(wk, wk16, 262144);
  cvt_f32_to_f16<<<1024, 256, 0, stream>>>(wv, wv16, 262144);
  cvt_f32_to_f16<<<1024, 256, 0, stream>>>(wo, wo16, 262144);

  // 3) Q/K/V projections: all batches merged, M = 16*1024 = 16384
  dim3 gp(512 / 128, 16384 / 128, 1);
  gemm_wmma_kernel<1, 0><<<gp, 256, 0, stream>>>(hf, wq16, q16, nullptr, bq, nullptr,
                                                 16384, 512, 512, 0, 0, 0, 1.0f);
  gemm_wmma_kernel<1, 0><<<gp, 256, 0, stream>>>(hf, wk16, k16, nullptr, bk, nullptr,
                                                 16384, 512, 512, 0, 0, 0, 1.0f);
  gemm_wmma_kernel<1, 0><<<gp, 256, 0, stream>>>(hf, wv16, v16, nullptr, bv, nullptr,
                                                 16384, 512, 512, 0, 0, 0, 1.0f);

  // 4) Scores: S = (Q K^T) * C^-0.5, per batch [1024,1024], K stored [N][K]
  dim3 gs(1024 / 128, 1024 / 128, 16);
  gemm_wmma_kernel<1, 0><<<gs, 256, 0, stream>>>(q16, k16, p16, nullptr, nullptr, nullptr,
                                                 1024, 1024, 512,
                                                 524288, 524288, 1048576,
                                                 0.044194173824159216f);

  // 5) Softmax in place on p16 rows
  softmax_kernel<<<16384, 256, 0, stream>>>(p16);

  // 6) O = P V, per batch: P[1024,1024] x V[1024,512] (B row-major [K][N])
  dim3 gv(512 / 128, 1024 / 128, 16);
  gemm_wmma_kernel<0, 0><<<gv, 256, 0, stream>>>(p16, v16, o16, nullptr, nullptr, nullptr,
                                                 1024, 512, 1024,
                                                 1048576, 524288, 524288, 1.0f);

  // 7) Final projection + bias + residual, transposed store to [B, C, H, W]
  dim3 gf(512 / 128, 16384 / 128, 1);
  gemm_wmma_kernel<1, 1><<<gf, 256, 0, stream>>>(o16, wo16, nullptr, out, bo, x,
                                                 16384, 512, 512, 0, 0, 0, 1.0f);
}
